// KGMultiHeadAttention_35261681500312
// MI455X (gfx1250) — matrix operand order repaired and checked by reading, not compiled, not run on hardware
//
#include <hip/hip_runtime.h>

// Multi-head attention forward, MI455X (gfx1250, wave32, WMMA f16 16x16x32).
// B=4, S=1024, D=1024, H=16, DK=64.
#define B_  4
#define S_  1024
#define D_  1024
#define H_  16
#define DK_ 64

typedef _Float16 half_t;
typedef __attribute__((ext_vector_type(16))) _Float16 v16h;
typedef __attribute__((ext_vector_type(8)))  _Float16 v8h;
typedef __attribute__((ext_vector_type(4)))  _Float16 v4h;
typedef __attribute__((ext_vector_type(8)))  float    v8f;

// ---------------------------------------------------------------------------
// Fragment loaders (CDNA5 WMMA 16-bit layouts, CDNA5 ISA §7.12.2)
// A (16x32, MxK): lane<16 -> row=lane, halves {K0..7, K16..23};
//                 lane>=16 -> row=lane-16, halves {K8..15, K24..31}.
// B (32x16, KxN): lane -> col n=lane&15, halves {K0..7, K16..23} (lane<16)
//                 or {K8..15, K24..31} (lane>=16), matching A.
// C/D (16x16 f32): VGPR r holds row r + 8*(lane>=16), col = lane&15.
// ---------------------------------------------------------------------------
static __device__ __forceinline__ v16h load_a_frag(const half_t* __restrict__ base,
                                                   int ld, int lane) {
  int m    = lane & 15;
  int koff = (lane >> 4) * 8;
  const half_t* p = base + (size_t)m * ld + koff;
  v8h lo = *(const v8h*)(p);
  v8h hi = *(const v8h*)(p + 16);
  v16h a;
#pragma unroll
  for (int i = 0; i < 8; ++i) { a[i] = lo[i]; a[i + 8] = hi[i]; }
  return a;
}

static __device__ __forceinline__ v16h load_b_frag(const half_t* __restrict__ base,
                                                   int ld, int lane) {
  const half_t* p = base + (size_t)(lane & 15) * ld + (lane >> 4) * 8;
  v8h lo = *(const v8h*)(p);
  v8h hi = *(const v8h*)(p + 16);
  v16h b;
#pragma unroll
  for (int i = 0; i < 8; ++i) { b[i] = lo[i]; b[i + 8] = hi[i]; }
  return b;
}

static __device__ __forceinline__ v8f wmma_f32_f16(v16h a, v16h b, v8f c) {
  return __builtin_amdgcn_wmma_f32_16x16x32_f16(false, a, false, b,
                                                (short)0, c, false, false);
}

// ---------------------------------------------------------------------------
// f32 -> f16 conversion (weights + activations), 4 elems/thread, b128 in / b64 out
// ---------------------------------------------------------------------------
__global__ __launch_bounds__(256) void cvt_f32_f16(const float* __restrict__ src,
                                                   half_t* __restrict__ dst, int n) {
  int i = (blockIdx.x * 256 + threadIdx.x) * 4;
  if (i < n) {
    float4 v = *(const float4*)(src + i);
    v4h o;
    o[0] = (half_t)v.x; o[1] = (half_t)v.y; o[2] = (half_t)v.z; o[3] = (half_t)v.w;
    *(v4h*)(dst + i) = o;
  }
}

// ---------------------------------------------------------------------------
// QKV projection: y[b,s,e] = sum_d x[b,s,d]*W[e,d] + bias[e].
// One wave computes a 16(M) x 64(N) tile; 64 N-cols == exactly one head, so the
// epilogue scatters into per-head layouts:
//   p=0 -> qh [B,H,S,DK], p=1 -> kh [B,H,S,DK], p=2 -> vt [B,H,DK,S] (transposed)
// ---------------------------------------------------------------------------
__global__ __launch_bounds__(32) void proj_qkv(
    const half_t* __restrict__ qf, const half_t* __restrict__ kf,
    const half_t* __restrict__ vf,
    const half_t* __restrict__ wq, const half_t* __restrict__ wk,
    const half_t* __restrict__ wv,
    const float* __restrict__ bq, const float* __restrict__ bk,
    const float* __restrict__ bvec,
    half_t* __restrict__ qh, half_t* __restrict__ kh, half_t* __restrict__ vt) {
  int lane = threadIdx.x;
  int p = blockIdx.z;
  const half_t* X  = (p == 0) ? qf : (p == 1) ? kf : vf;
  const half_t* Wm = (p == 0) ? wq : (p == 1) ? wk : wv;
  const float*  bb = (p == 0) ? bq : (p == 1) ? bk : bvec;
  int m0 = blockIdx.y * 16;   // row tile over M = B*S
  int h  = blockIdx.x;        // head == 64-col N tile
  int n0 = h * 64;

  v8f acc[4] = {};
  for (int kc = 0; kc < D_; kc += 32) {
    v16h a = load_a_frag(X + (size_t)m0 * D_ + kc, D_, lane);
#pragma unroll
    for (int t = 0; t < 4; ++t) {
      v16h bf = load_b_frag(Wm + (size_t)(n0 + t * 16) * D_ + kc, D_, lane);
      acc[t] = wmma_f32_f16(a, bf, acc[t]);
    }
  }

  int hi = lane >> 4, nl = lane & 15;
#pragma unroll
  for (int t = 0; t < 4; ++t) {
    float badd = bb[n0 + t * 16 + nl];
    int dk = t * 16 + nl;
#pragma unroll
    for (int r = 0; r < 8; ++r) {
      int m = m0 + r + 8 * hi;
      int b = m >> 10, s = m & (S_ - 1);
      half_t val = (half_t)(acc[t][r] + badd);
      if (p == 2) {
        vt[((size_t)(b * H_ + h) * DK_ + dk) * S_ + s] = val;
      } else {
        half_t* dst = (p == 0) ? qh : kh;
        dst[((size_t)(b * H_ + h) * S_ + s) * DK_ + dk] = val;
      }
    }
  }
}

// ---------------------------------------------------------------------------
// Flash attention: one wave per (b, h, 16-query tile). Streams bias exactly
// once, fused scale+bias+mask+online-softmax, PV accumulated with WMMA.
// ---------------------------------------------------------------------------
__global__ __launch_bounds__(32) void flash_attn(
    const half_t* __restrict__ qh, const half_t* __restrict__ kh,
    const half_t* __restrict__ vt,
    const float* __restrict__ bias, const int* __restrict__ mask,
    half_t* __restrict__ xh) {
  __shared__ __align__(16) half_t Plds[16][40];  // padded: 80B rows, no 4-way bank camp

  int lane = threadIdx.x;
  int q0 = blockIdx.x * 16, h = blockIdx.y, b = blockIdx.z;
  const half_t* Q  = qh + (size_t)(b * H_ + h) * S_ * DK_;
  const half_t* K  = kh + (size_t)(b * H_ + h) * S_ * DK_;
  const half_t* V  = vt + (size_t)(b * H_ + h) * DK_ * S_;
  const float*  Bp = bias + ((size_t)(b * H_ + h) * S_ + q0) * S_;
  const int*    Mp = mask + ((size_t)b * S_ + q0) * S_;

  // Q fragments (held for whole loop): 16 rows x K=64 as two 16x32 A-frags
  v16h qa0 = load_a_frag(Q + (size_t)q0 * DK_ + 0,  DK_, lane);
  v16h qa1 = load_a_frag(Q + (size_t)q0 * DK_ + 32, DK_, lane);

  v8f acc[4] = {};
  float mrow[8], lrow[8];
#pragma unroll
  for (int r = 0; r < 8; ++r) { mrow[r] = -1e30f; lrow[r] = 0.0f; }
  int hi = lane >> 4, nl = lane & 15;

  for (int k0 = 0; k0 < S_; k0 += 32) {
    // keep the dominant 256MB bias stream ahead of compute
    if (k0 + 32 < S_) {
#pragma unroll
      for (int r = 0; r < 8; ++r)
        __builtin_prefetch(Bp + (size_t)(r + 8 * hi) * S_ + k0 + 32 + nl, 0, 1);
    }

    // scores: two 16x16 tiles (keys k0..k0+15 and k0+16..k0+31), K=64 = 2 chunks
    v8f s0 = {}, s1 = {};
    {
      v16h kb0 = load_b_frag(K + (size_t)(k0)      * DK_ + 0,  DK_, lane);
      v16h kb1 = load_b_frag(K + (size_t)(k0)      * DK_ + 32, DK_, lane);
      s0 = wmma_f32_f16(qa0, kb0, s0);
      s0 = wmma_f32_f16(qa1, kb1, s0);
      v16h kc0 = load_b_frag(K + (size_t)(k0 + 16) * DK_ + 0,  DK_, lane);
      v16h kc1 = load_b_frag(K + (size_t)(k0 + 16) * DK_ + 32, DK_, lane);
      s1 = wmma_f32_f16(qa0, kc0, s1);
      s1 = wmma_f32_f16(qa1, kc1, s1);
    }

    // fused scale (1/sqrt(64)) + bias + mask; coalesced b32 loads across lanes
#pragma unroll
    for (int r = 0; r < 8; ++r) {
      int row = r + 8 * hi;
      float b0 = Bp[(size_t)row * S_ + k0 + nl];
      float b1 = Bp[(size_t)row * S_ + k0 + 16 + nl];
      int   z0 = Mp[(size_t)row * S_ + k0 + nl];
      int   z1 = Mp[(size_t)row * S_ + k0 + 16 + nl];
      float x0 = s0[r] * 0.125f + b0;
      float x1 = s1[r] * 0.125f + b1;
      s0[r] = z0 ? x0 : -1e30f;
      s1[r] = z1 ? x1 : -1e30f;
    }

    // online softmax; row reductions are width-16 xor-shuffles (stay in half-wave)
#pragma unroll
    for (int r = 0; r < 8; ++r) {
      float v = fmaxf(s0[r], s1[r]);
#pragma unroll
      for (int off = 8; off >= 1; off >>= 1) v = fmaxf(v, __shfl_xor(v, off, 32));
      float mn   = fmaxf(mrow[r], v);
      float corr = __expf(mrow[r] - mn);
      float p0   = __expf(s0[r] - mn);
      float p1   = __expf(s1[r] - mn);
      float sum  = p0 + p1;
#pragma unroll
      for (int off = 8; off >= 1; off >>= 1) sum += __shfl_xor(sum, off, 32);
      lrow[r] = lrow[r] * corr + sum;
      mrow[r] = mn;
      s0[r] = p0; s1[r] = p1;
#pragma unroll
      for (int t = 0; t < 4; ++t) acc[t][r] *= corr;
    }

    // re-layout P (C-layout -> A-layout) via 1.25KB LDS
#pragma unroll
    for (int r = 0; r < 8; ++r) {
      Plds[r + 8 * hi][nl]      = (half_t)s0[r];
      Plds[r + 8 * hi][nl + 16] = (half_t)s1[r];
    }
    __syncthreads();
    v16h pa;
    {
      const half_t* p = &Plds[nl][hi * 8];
      v8h lo = *(const v8h*)p;
      v8h hh = *(const v8h*)(p + 16);
#pragma unroll
      for (int i = 0; i < 8; ++i) { pa[i] = lo[i]; pa[i + 8] = hh[i]; }
    }
    __syncthreads();

    // PV: A = P (16x32), B = V^T slices (contiguous along keys), 4 dk tiles
#pragma unroll
    for (int t = 0; t < 4; ++t) {
      v16h vb = load_b_frag(V + (size_t)(t * 16) * S_ + k0, S_, lane);
      acc[t] = wmma_f32_f16(pa, vb, acc[t]);
    }
  }

  // normalize and write x as [B,S,D] f16 (heads re-concatenated)
#pragma unroll
  for (int r = 0; r < 8; ++r) {
    float inv = 1.0f / lrow[r];
    int srow = q0 + r + 8 * hi;
#pragma unroll
    for (int t = 0; t < 4; ++t) {
      xh[((size_t)(b * S_ + srow)) * D_ + h * DK_ + t * 16 + nl] =
          (half_t)(acc[t][r] * inv);
    }
  }
}

// ---------------------------------------------------------------------------
// Output projection: out[b,s,e] = sum_d x[b,s,d]*Wo[e,d] + bo[e]  (f32 out)
// ---------------------------------------------------------------------------
__global__ __launch_bounds__(32) void proj_out(
    const half_t* __restrict__ xh, const half_t* __restrict__ wo,
    const float* __restrict__ bo, float* __restrict__ out) {
  int lane = threadIdx.x;
  int m0 = blockIdx.y * 16, n0 = blockIdx.x * 64;
  v8f acc[4] = {};
  for (int kc = 0; kc < D_; kc += 32) {
    v16h a = load_a_frag(xh + (size_t)m0 * D_ + kc, D_, lane);
#pragma unroll
    for (int t = 0; t < 4; ++t) {
      v16h bf = load_b_frag(wo + (size_t)(n0 + t * 16) * D_ + kc, D_, lane);
      acc[t] = wmma_f32_f16(a, bf, acc[t]);
    }
  }
  int hi = lane >> 4, nl = lane & 15;
#pragma unroll
  for (int t = 0; t < 4; ++t) {
    float badd = bo[n0 + t * 16 + nl];
#pragma unroll
    for (int r = 0; r < 8; ++r) {
      int m = m0 + r + 8 * hi;
      out[(size_t)m * D_ + n0 + t * 16 + nl] = acc[t][r] + badd;
    }
  }
}

// ---------------------------------------------------------------------------
// Host launcher
// ---------------------------------------------------------------------------
extern "C" void kernel_launch(void* const* d_in, const int* in_sizes, int n_in,
                              void* d_out, int out_size, void* d_ws, size_t ws_size,
                              hipStream_t stream) {
  const float* query = (const float*)d_in[0];
  const float* key_  = (const float*)d_in[1];
  const float* value = (const float*)d_in[2];
  const int*   mask  = (const int*)  d_in[3];
  const float* bias  = (const float*)d_in[4];
  const float* Wq = (const float*)d_in[5];
  const float* bq = (const float*)d_in[6];
  const float* Wk = (const float*)d_in[7];
  const float* bk = (const float*)d_in[8];
  const float* Wv = (const float*)d_in[9];
  const float* bv = (const float*)d_in[10];
  const float* Wo = (const float*)d_in[11];
  const float* bo = (const float*)d_in[12];
  float* out = (float*)d_out;

  // workspace layout (f16 elements): 3x act(4M) + 4x weight(1M) + q/k/vT/x(4x4M) = 64MB
  half_t* W = (half_t*)d_ws;
  const size_t M4 = (size_t)B_ * S_ * D_;   // 4,194,304
  const size_t M1 = (size_t)D_ * D_;        // 1,048,576
  half_t* qf = W;
  half_t* kf = qf + M4;
  half_t* vf = kf + M4;
  half_t* wq = vf + M4;
  half_t* wk = wq + M1;
  half_t* wv = wk + M1;
  half_t* wo = wv + M1;
  half_t* qh = wo + M1;
  half_t* kh = qh + M4;
  half_t* vt = kh + M4;
  half_t* xh = vt + M4;

  // 1) convert inputs + weights to f16
  cvt_f32_f16<<<dim3((unsigned)(M4 / 4 / 256)), dim3(256), 0, stream>>>(query, qf, (int)M4);
  cvt_f32_f16<<<dim3((unsigned)(M4 / 4 / 256)), dim3(256), 0, stream>>>(key_,  kf, (int)M4);
  cvt_f32_f16<<<dim3((unsigned)(M4 / 4 / 256)), dim3(256), 0, stream>>>(value, vf, (int)M4);
  cvt_f32_f16<<<dim3((unsigned)(M1 / 4 / 256)), dim3(256), 0, stream>>>(Wq, wq, (int)M1);
  cvt_f32_f16<<<dim3((unsigned)(M1 / 4 / 256)), dim3(256), 0, stream>>>(Wk, wk, (int)M1);
  cvt_f32_f16<<<dim3((unsigned)(M1 / 4 / 256)), dim3(256), 0, stream>>>(Wv, wv, (int)M1);
  cvt_f32_f16<<<dim3((unsigned)(M1 / 4 / 256)), dim3(256), 0, stream>>>(Wo, wo, (int)M1);

  // 2) fused QKV projections (z = q/k/v)
  proj_qkv<<<dim3(H_, (B_ * S_) / 16, 3), dim3(32), 0, stream>>>(
      qf, kf, vf, wq, wk, wv, bq, bk, bv, qh, kh, vt);

  // 3) flash attention (bias/mask/softmax fused, bias read exactly once)
  flash_attn<<<dim3(S_ / 16, H_, B_), dim3(32), 0, stream>>>(
      qh, kh, vt, bias, mask, xh);

  // 4) output projection -> f32 d_out
  proj_out<<<dim3(H_, (B_ * S_) / 16), dim3(32), 0, stream>>>(xh, wo, bo, out);
}
